// IterGruAutoencoder_64604898067063
// MI455X (gfx1250) — matrix operand order repaired and checked by reading, not compiled, not run on hardware
//
#include <hip/hip_runtime.h>
#include <hip/hip_bf16.h>

// ---------------------------------------------------------------------------
// GRU autoencoder, persistent-kernel formulation for MI455X (gfx1250).
//  - 4 workgroups x 256 threads (8 wave32s); WG g owns batch rows 16g..16g+15.
//  - Weights fp32 -> f16 into LDS; hidden state fp32 in VGPRs (8 per lane).
//  - Wave w owns output columns 16w..16w+15. Since the GRU gates for column c
//    need gi/gh columns {c, 128+c, 256+c} = tiles {w, w+8, w+16}, the whole
//    gate computation runs on WMMA accumulators in registers: no gi/gh LDS
//    round trip at all. Only the f16 hidden copy (WMMA A operand) hits LDS.
//  - 6-7 independent accumulators per wave interleave WMMAs (hides D->C RAW).
//  - Decoder: x == h, so 4 A-fragments feed out-proj + gi + gh (28 wmma/step).
// ---------------------------------------------------------------------------

typedef __attribute__((ext_vector_type(16))) _Float16 v16h;
typedef __attribute__((ext_vector_type(8)))  _Float16 v8h;
typedef __attribute__((ext_vector_type(8)))  float    v8f;

#define Bsz 64
#define Tsz 2048
#define Fsz 128
#define Hsz 128
#define G3  384
#define ROWS 16
#define NTHREADS 256
#define NWAVES 8

#define WMMA(a, b, c) \
  __builtin_amdgcn_wmma_f32_16x16x32_f16(false, (a), false, (b), (short)0, (c), false, false)

// LDS: w0 384x128 f16, w1 384x128 f16, w2 128x128 f16, sX 16x128 f16, sH16 16x128 f16
static constexpr size_t LDS_BYTES =
    (size_t)(2 * G3 * Hsz + Hsz * Hsz + 2 * ROWS * Hsz) * 2;

// ---- WMMA fragment loaders (contiguous k = 16h + s in A and B; row-major f16 LDS) --------

// A tile 16x32 at `base` (k-origin folded in), ld = Hsz.
__device__ __forceinline__ v16h afrag(const _Float16* __restrict__ base, int lane) {
  const _Float16* p = base + (lane & 15) * Hsz + ((lane & 16) ? 16 : 0);
  v8h lo = *(const v8h*)(p);        // K = kb .. kb+7
  v8h hi = *(const v8h*)(p + 8);    // K = kb+8 .. kb+15
  v16h a;
#pragma unroll
  for (int i = 0; i < 8; ++i) { a[i] = lo[i]; a[8 + i] = hi[i]; }
  return a;
}

// B tile 32x16 where B = W^T, W row-major [N x 128].
__device__ __forceinline__ v16h bfrag(const _Float16* __restrict__ W,
                                      int n0, int k0, int lane) {
  return *(const v16h*)(W + (n0 + (lane & 15)) * Hsz + k0 + ((lane & 16) ? 16 : 0));
}

__device__ __forceinline__ void loadw(const float* __restrict__ s,
                                      _Float16* __restrict__ d, int n, int tid) {
  for (int i = tid; i < n; i += NTHREADS) d[i] = (_Float16)s[i];
}

__device__ __forceinline__ float sigm(float x) { return 1.f / (1.f + __expf(-x)); }

extern "C" __global__ __launch_bounds__(NTHREADS)
void gru_autoenc_kernel(const float* __restrict__ to_x,
                        const int* __restrict__ lengths,
                        const float* __restrict__ eWih, const float* __restrict__ eWhh,
                        const float* __restrict__ ebih, const float* __restrict__ ebhh,
                        const float* __restrict__ fcW,  const float* __restrict__ fcb,
                        const float* __restrict__ dWih, const float* __restrict__ dWhh,
                        const float* __restrict__ dbih, const float* __restrict__ dbhh,
                        const float* __restrict__ oW,   const float* __restrict__ ob,
                        float* __restrict__ out) {
  extern __shared__ unsigned char smem[];
  _Float16* w0   = (_Float16*)smem;        // 384x128: enc_Wih / fc_W / dec_Wih
  _Float16* w1   = w0 + G3 * Hsz;          // 384x128: enc_Whh / dec_Whh
  _Float16* w2   = w1 + G3 * Hsz;          // 128x128: out_W
  _Float16* sX   = w2 + Hsz * Hsz;         // 16x128 : x_t / encoded (f16 A)
  _Float16* sH16 = sX + ROWS * Hsz;        // 16x128 : hidden (f16 A)

  const int tid    = threadIdx.x;
  const int lane   = tid & 31;
  const int wave   = tid >> 5;
  const int wgBase = blockIdx.x * ROWS;

  const int col = wave * 16 + (lane & 15);   // output column owned by this lane
  const int rb  = (lane & 16) ? 8 : 0;       // accumulator row base (C layout)
  const int n0  = wave * 16;                 // first N-tile row offset
  const int n1  = (wave + 8) * 16;
  const int n2  = (wave + 16) * 16;

  // per-lane persistent state: fp32 hidden for rows rb..rb+7 at column `col`
  float hf[8];
#pragma unroll
  for (int v = 0; v < 8; ++v) hf[v] = 0.f;
  int lenv[8];
#pragma unroll
  for (int v = 0; v < 8; ++v) lenv[v] = lengths[wgBase + rb + v];

  // encoder gate biases (per column -> per lane registers)
  float bi0 = ebih[col], bi1 = ebih[Hsz + col], bi2 = ebih[2 * Hsz + col];
  float bh0 = ebhh[col], bh1 = ebhh[Hsz + col], bh2 = ebhh[2 * Hsz + col];

  // ---------------- prologue: encoder weights, h = 0 ----------------------
  loadw(eWih, w0, G3 * Hsz, tid);
  loadw(eWhh, w1, G3 * Hsz, tid);
  for (int i = tid; i < ROWS * Hsz; i += NTHREADS) sH16[i] = (_Float16)0.f;
  __syncthreads();

  // ---------------- encoder scan ------------------------------------------
  for (int t = 0; t < Tsz; ++t) {
    // phase W: stage x_t (and publish h from previous step)
    for (int i = tid; i < ROWS * Hsz; i += NTHREADS) {
      int r = i >> 7, c = i & (Hsz - 1);
      sX[i] = (_Float16)to_x[((size_t)(wgBase + r) * Tsz + t) * Fsz + c];
    }
    if (t > 0) {
#pragma unroll
      for (int v = 0; v < 8; ++v) sH16[(rb + v) * Hsz + col] = (_Float16)hf[v];
    }
    if (t + 1 < Tsz) {  // prefetch next timestep tile (global_prefetch_b8)
      for (int i = tid; i < ROWS * 4; i += NTHREADS) {
        int r = i >> 2, seg = i & 3;
        __builtin_prefetch(
            to_x + ((size_t)(wgBase + r) * Tsz + (t + 1)) * Fsz + seg * 32, 0, 0);
      }
    }
    __syncthreads();

    // phase R: dual GEMM, 6 independent accumulators, gates in registers
    v8f ci0 = {}, ci1 = {}, ci2 = {}, ch0 = {}, ch1 = {}, ch2 = {};
#pragma unroll
    for (int kc = 0; kc < 4; ++kc) {
      v16h aX = afrag(sX + kc * 32, lane);
      v16h aH = afrag(sH16 + kc * 32, lane);
      ci0 = WMMA(aX, bfrag(w0, n0, kc * 32, lane), ci0);
      ci1 = WMMA(aX, bfrag(w0, n1, kc * 32, lane), ci1);
      ci2 = WMMA(aX, bfrag(w0, n2, kc * 32, lane), ci2);
      ch0 = WMMA(aH, bfrag(w1, n0, kc * 32, lane), ch0);
      ch1 = WMMA(aH, bfrag(w1, n1, kc * 32, lane), ch1);
      ch2 = WMMA(aH, bfrag(w1, n2, kc * 32, lane), ch2);
    }
#pragma unroll
    for (int v = 0; v < 8; ++v) {
      float rg = sigm(ci0[v] + bi0 + ch0[v] + bh0);
      float zg = sigm(ci1[v] + bi1 + ch1[v] + bh1);
      float nn = tanhf(ci2[v] + bi2 + rg * (ch2[v] + bh2));
      float hnew = (1.f - zg) * nn + zg * hf[v];
      hf[v] = (t < lenv[v]) ? hnew : hf[v];   // packed-seq freeze
    }
    __syncthreads();
  }

  // publish h_enc (f16)
#pragma unroll
  for (int v = 0; v < 8; ++v) sH16[(rb + v) * Hsz + col] = (_Float16)hf[v];
  loadw(fcW, w0, Hsz * Hsz, tid);   // reuse w0 region (reads done at barrier)
  float fcbv = fcb[col];
  __syncthreads();

  // ---------------- encoded = h_enc @ fc_W^T + fc_b (8 tiles, 1/wave) -----
  {
    v8f cf = {};
#pragma unroll
    for (int kc = 0; kc < 4; ++kc)
      cf = WMMA(afrag(sH16 + kc * 32, lane), bfrag(w0, n0, kc * 32, lane), cf);
    __syncthreads();   // fc_W / sH16 reads done
#pragma unroll
    for (int v = 0; v < 8; ++v)
      sX[(rb + v) * Hsz + col] = (_Float16)(cf[v] + fcbv);
  }
  // decoder weights + biases
  loadw(dWih, w0, G3 * Hsz, tid);
  loadw(dWhh, w1, G3 * Hsz, tid);
  loadw(oW,  w2, Hsz * Hsz, tid);
  bi0 = dbih[col]; bi1 = dbih[Hsz + col]; bi2 = dbih[2 * Hsz + col];
  bh0 = dbhh[col]; bh1 = dbhh[Hsz + col]; bh2 = dbhh[2 * Hsz + col];
  float ov = ob[col];
  __syncthreads();

  // ---------------- y0 = dec_cell(encoded, h_enc) -------------------------
  {
    v8f ci0 = {}, ci1 = {}, ci2 = {}, ch0 = {}, ch1 = {}, ch2 = {};
#pragma unroll
    for (int kc = 0; kc < 4; ++kc) {
      v16h aX = afrag(sX + kc * 32, lane);    // encoded
      v16h aH = afrag(sH16 + kc * 32, lane);  // h_enc
      ci0 = WMMA(aX, bfrag(w0, n0, kc * 32, lane), ci0);
      ci1 = WMMA(aX, bfrag(w0, n1, kc * 32, lane), ci1);
      ci2 = WMMA(aX, bfrag(w0, n2, kc * 32, lane), ci2);
      ch0 = WMMA(aH, bfrag(w1, n0, kc * 32, lane), ch0);
      ch1 = WMMA(aH, bfrag(w1, n1, kc * 32, lane), ch1);
      ch2 = WMMA(aH, bfrag(w1, n2, kc * 32, lane), ch2);
    }
#pragma unroll
    for (int v = 0; v < 8; ++v) {
      float rg = sigm(ci0[v] + bi0 + ch0[v] + bh0);
      float zg = sigm(ci1[v] + bi1 + ch1[v] + bh1);
      float nn = tanhf(ci2[v] + bi2 + rg * (ch2[v] + bh2));
      hf[v] = (1.f - zg) * nn + zg * hf[v];   // hf now holds y
    }
    __syncthreads();
#pragma unroll
    for (int v = 0; v < 8; ++v) sH16[(rb + v) * Hsz + col] = (_Float16)hf[v];
    __syncthreads();
  }

  // ---------------- decoder loop: emit out[t]; y <- cell(y, y) ------------
  for (int t = 0; t < Tsz; ++t) {
    const bool more = (t + 1 < Tsz);
    v8f co = {}, ci0 = {}, ci1 = {}, ci2 = {}, ch0 = {}, ch1 = {}, ch2 = {};
#pragma unroll
    for (int kc = 0; kc < 4; ++kc) {
      v16h a = afrag(sH16 + kc * 32, lane);   // y_t feeds out-proj AND gi/gh
      co = WMMA(a, bfrag(w2, n0, kc * 32, lane), co);
      if (more) {
        ci0 = WMMA(a, bfrag(w0, n0, kc * 32, lane), ci0);
        ci1 = WMMA(a, bfrag(w0, n1, kc * 32, lane), ci1);
        ci2 = WMMA(a, bfrag(w0, n2, kc * 32, lane), ci2);
        ch0 = WMMA(a, bfrag(w1, n0, kc * 32, lane), ch0);
        ch1 = WMMA(a, bfrag(w1, n1, kc * 32, lane), ch1);
        ch2 = WMMA(a, bfrag(w1, n2, kc * 32, lane), ch2);
      }
    }
    // out[b, t, col] = co + out_b   (direct from accumulators)
#pragma unroll
    for (int v = 0; v < 8; ++v)
      out[(size_t)(wgBase + rb + v) * (Tsz * Hsz) + (size_t)t * Hsz + col] =
          co[v] + ov;
    if (more) {
#pragma unroll
      for (int v = 0; v < 8; ++v) {
        float rg = sigm(ci0[v] + bi0 + ch0[v] + bh0);
        float zg = sigm(ci1[v] + bi1 + ch1[v] + bh1);
        float nn = tanhf(ci2[v] + bi2 + rg * (ch2[v] + bh2));
        hf[v] = (1.f - zg) * nn + zg * hf[v];
      }
      __syncthreads();   // all reads of sH16 done
#pragma unroll
      for (int v = 0; v < 8; ++v) sH16[(rb + v) * Hsz + col] = (_Float16)hf[v];
      __syncthreads();   // new y visible
    }
  }
}

extern "C" void kernel_launch(void* const* d_in, const int* in_sizes, int n_in,
                              void* d_out, int out_size, void* d_ws, size_t ws_size,
                              hipStream_t stream) {
  (void)in_sizes; (void)n_in; (void)d_ws; (void)ws_size; (void)out_size;
  const float* to_x    = (const float*)d_in[0];
  const int*   lengths = (const int*)  d_in[1];
  const float* eWih    = (const float*)d_in[2];
  const float* eWhh    = (const float*)d_in[3];
  const float* ebih    = (const float*)d_in[4];
  const float* ebhh    = (const float*)d_in[5];
  const float* fcW     = (const float*)d_in[6];
  const float* fcb     = (const float*)d_in[7];
  const float* dWih    = (const float*)d_in[8];
  const float* dWhh    = (const float*)d_in[9];
  const float* dbih    = (const float*)d_in[10];
  const float* dbhh    = (const float*)d_in[11];
  const float* oW      = (const float*)d_in[12];
  const float* ob      = (const float*)d_in[13];
  float* out = (float*)d_out;

  dim3 grid(Bsz / ROWS);     // 4 persistent workgroups, one per 16 batch rows
  dim3 block(NTHREADS);      // 8 wave32s
  gru_autoenc_kernel<<<grid, block, LDS_BYTES, stream>>>(
      to_x, lengths, eWih, eWhh, ebih, ebhh, fcW, fcb,
      dWih, dWhh, dbih, dbhh, oW, ob, out);
}
